// MinibatchDiscrimination_33483565040179
// MI455X (gfx1250) — matrix operand order repaired and checked by reading, not compiled, not run on hardware
//
#include <hip/hip_runtime.h>
#include <hip/hip_bf16.h>

typedef __attribute__((ext_vector_type(16))) _Float16 v16h;
typedef __attribute__((ext_vector_type(8)))  _Float16 v8h;
typedef __attribute__((ext_vector_type(8)))  float    v8f;

#define B_    512
#define IN_   1024
#define OUT_  128
#define K_    8
#define NCOL  1024           // OUT_ * K_
#define OUTW  1152           // IN_ + OUT_

// ---------------------------------------------------------------------------
// Elementwise f32 -> f16 conversion, 8 elements per thread (2x float4 in,
// one 16B half8 out). Run once for x and once for T; amortized across the
// 32-64x reuse of those operands in the GEMM.
// ---------------------------------------------------------------------------
__global__ __launch_bounds__(256)
void mbd_f32_to_f16_kernel(const float* __restrict__ src,
                           _Float16* __restrict__ dst) {
  const int idx = blockIdx.x * 256 + threadIdx.x;       // one half8 per thread
  const float4* s = reinterpret_cast<const float4*>(src) + idx * 2;
  const float4 f0 = s[0], f1 = s[1];
  v8h h;
  h[0]=(_Float16)f0.x; h[1]=(_Float16)f0.y; h[2]=(_Float16)f0.z; h[3]=(_Float16)f0.w;
  h[4]=(_Float16)f1.x; h[5]=(_Float16)f1.y; h[6]=(_Float16)f1.z; h[7]=(_Float16)f1.w;
  reinterpret_cast<v8h*>(dst)[idx] = h;
}

// ---------------------------------------------------------------------------
// GEMM: M(512x1024) = xh(512x1024,f16) @ Th(1024x1024,f16), f32 accumulate.
// One wave computes a 16x64 tile: one A fragment feeds 4 independent
// v_wmma_f32_16x16x32_f16 accumulator chains (4x A reuse, ILP for the XDL
// pipe). K-loop unrolled 2x -> 8 WMMAs per loop body, no cvt in the loop.
//
// A layout (16-bit A 16x32): lane L (m = L&15):
//   L<16 : elems 0..7 = K 0..7,   elems 8..15 = K 16..23
//   L>=16: elems 0..7 = K 8..15,  elems 8..15 = K 24..31
// B layout (32x16): lane L holds col N=L&15, elems 0..15 = K in the same order as A.
// C/D layout: VGPR r: lanes 0-15 -> (row r, col lane); lanes 16-31 -> (row r+8).
// ---------------------------------------------------------------------------
__global__ __launch_bounds__(256)
void mbd_gemm_wmma_kernel(const _Float16* __restrict__ xh,
                          const _Float16* __restrict__ Th,
                          float* __restrict__ M) {
  const int tid   = threadIdx.x;
  const int lane  = tid & 31;
  const int wave  = tid >> 5;
  const int tile  = blockIdx.x * 8 + wave;   // 512 tiles = 32 m-tiles x 16 n-supertiles
  const int nst   = tile & 15;               // 64-wide n supertile
  const int mtile = tile >> 4;

  const int m   = lane & 15;
  const int khi = (lane >> 4) << 3;          // 0 for lanes 0-15, 8 for lanes 16-31

  const _Float16* arow  = xh + (mtile * 16 + m) * IN_ + khi;
  const _Float16* bbase = Th + nst * 64;

  v8f acc0 = {}, acc1 = {}, acc2 = {}, acc3 = {};
#pragma unroll 2
  for (int kk = 0; kk < IN_; kk += 32) {
    // A fragment: two 16B loads per lane (K base..base+7, base+16..base+23)
    const v8h alo = *reinterpret_cast<const v8h*>(arow + kk);
    const v8h ahi = *reinterpret_cast<const v8h*>(arow + kk + 16);
    const v16h a = __builtin_shufflevector(alo, ahi,
        0,1,2,3,4,5,6,7,8,9,10,11,12,13,14,15);
    // B fragments: lane holds column N=lane&15; elems follow A's K order (4 n-subtiles)
    const _Float16* bcol = bbase + (kk + khi) * NCOL + (lane & 15);
    v16h b0, b1, b2, b3;
#pragma unroll
    for (int s = 0; s < 16; ++s) {
      const _Float16* bp = bcol + (s + ((s >> 3) << 3)) * NCOL;
      b0[s] = bp[0]; b1[s] = bp[16]; b2[s] = bp[32]; b3[s] = bp[48];
    }

    acc0 = __builtin_amdgcn_wmma_f32_16x16x32_f16(false, a, false, b0, (short)0, acc0, false, false);
    acc1 = __builtin_amdgcn_wmma_f32_16x16x32_f16(false, a, false, b1, (short)0, acc1, false, false);
    acc2 = __builtin_amdgcn_wmma_f32_16x16x32_f16(false, a, false, b2, (short)0, acc2, false, false);
    acc3 = __builtin_amdgcn_wmma_f32_16x16x32_f16(false, a, false, b3, (short)0, acc3, false, false);
  }

  // ---- store 16x64 f32 tile ----
  const int row0 = mtile * 16 + ((lane >> 4) << 3);   // +8 for upper lane half
  const int col  = nst * 64 + (lane & 15);
  float* mp = M + row0 * NCOL + col;
#pragma unroll
  for (int r = 0; r < 8; ++r) {
    mp[r * NCOL +  0] = acc0[r];
    mp[r * NCOL + 16] = acc1[r];
    mp[r * NCOL + 32] = acc2[r];
    mp[r * NCOL + 48] = acc3[r];
  }
}

// ---------------------------------------------------------------------------
// Pairwise: out[j, 1024+o] = sum_i exp(-sum_k |M[i,o,k]-M[j,o,k]|) - 1
// One block per (o, j-chunk of 256). Stage all 512 rows (16KB) of M[:,o,:]
// in LDS once; every thread streams them (LDS broadcast across the wave).
// ---------------------------------------------------------------------------
__global__ __launch_bounds__(256)
void mbd_pairwise_kernel(const float* __restrict__ M, float* __restrict__ out) {
  __shared__ float4 s[B_][2];                 // 512 x 8 floats = 16 KB
  const int o = blockIdx.x;                   // 0..127
  const int j = blockIdx.y * 256 + threadIdx.x;

  for (int idx = threadIdx.x; idx < B_ * 2; idx += 256) {
    const int i = idx >> 1, h = idx & 1;
    s[i][h] = reinterpret_cast<const float4*>(M + i * NCOL + o * K_)[h];
  }
  __syncthreads();

  const float4 mj0 = s[j][0], mj1 = s[j][1];
  float acc = 0.0f;
#pragma unroll 4
  for (int i = 0; i < B_; ++i) {
    const float4 a = s[i][0], b = s[i][1];
    float l1 = fabsf(a.x - mj0.x) + fabsf(a.y - mj0.y)
             + fabsf(a.z - mj0.z) + fabsf(a.w - mj0.w)
             + fabsf(b.x - mj1.x) + fabsf(b.y - mj1.y)
             + fabsf(b.z - mj1.z) + fabsf(b.w - mj1.w);
    acc += __expf(-l1);
  }
  out[j * OUTW + IN_ + o] = acc - 1.0f;
}

// ---------------------------------------------------------------------------
// Copy x -> out[:, :1024] as float4 (both row bases 16B-aligned).
// ---------------------------------------------------------------------------
__global__ __launch_bounds__(256)
void mbd_copyx_kernel(const float* __restrict__ x, float* __restrict__ out) {
  const int idx = blockIdx.x * 256 + threadIdx.x;   // 512*256 float4s
  const int row = idx >> 8;
  const int c4  = idx & 255;
  reinterpret_cast<float4*>(out + row * OUTW)[c4] =
      reinterpret_cast<const float4*>(x + row * IN_)[c4];
}

extern "C" void kernel_launch(void* const* d_in, const int* in_sizes, int n_in,
                              void* d_out, int out_size, void* d_ws, size_t ws_size,
                              hipStream_t stream) {
  const float* x = (const float*)d_in[0];   // (512, 1024)
  const float* T = (const float*)d_in[1];   // (1024, 128, 8) == (1024, 1024) row-major
  float* out = (float*)d_out;               // (512, 1152)

  // Workspace layout: xh (1MB) | Th (2MB) | M f32 (2MB)  => 5MB total
  char* ws = (char*)d_ws;
  _Float16* xh = (_Float16*)(ws);
  _Float16* Th = (_Float16*)(ws + (size_t)B_ * IN_ * sizeof(_Float16));
  float*    M  = (float*)   (ws + (size_t)B_ * IN_ * sizeof(_Float16)
                                + (size_t)IN_ * NCOL * sizeof(_Float16));

  // Pre-convert operands to f16 (amortized over 32-64x reuse in GEMM).
  mbd_f32_to_f16_kernel<<<(B_ * IN_) / 8 / 256, 256, 0, stream>>>(x, xh);   // 256 blocks
  mbd_f32_to_f16_kernel<<<(IN_ * NCOL) / 8 / 256, 256, 0, stream>>>(T, Th); // 512 blocks

  // 512 wave-tiles (16x64 each), 8 waves per 256-thread block -> 64 blocks
  mbd_gemm_wmma_kernel<<<64, 256, 0, stream>>>(xh, Th, M);

  // independent of GEMM result
  mbd_copyx_kernel<<<512, 256, 0, stream>>>(x, out);

  // consumes M (same stream -> ordered after GEMM)
  mbd_pairwise_kernel<<<dim3(OUT_, 2), 256, 0, stream>>>(M, out);
}
